// MultiHeadedCrossAttention_21036749815912
// MI455X (gfx1250) — matrix operand order repaired and checked by reading, not compiled, not run on hardware
//
#include <hip/hip_runtime.h>
#include <hip/hip_bf16.h>
#include <cstdint>

typedef _Float16     v16h __attribute__((ext_vector_type(16)));
typedef float        v8f  __attribute__((ext_vector_type(8)));
typedef unsigned int v4u  __attribute__((ext_vector_type(4)));
typedef int          v4i  __attribute__((ext_vector_type(4)));
typedef int          v8i  __attribute__((ext_vector_type(8)));

#define HIDDEN 1024
#define HEADS  16
#define DK     64
#define LQ_    1024
#define LKV_   2048

#if defined(__has_builtin)
#if __has_builtin(__builtin_amdgcn_tensor_load_to_lds)
#define USE_TDM 1
#endif
#endif
#ifndef USE_TDM
#define USE_TDM 0
#endif

#if __has_include(<hip/amd_detail/amd_gfx1250_TDM.h>)
#define TDM_6ARG 1     // therock-10.0 headers -> clang-23 6-arg builtin
#else
#define TDM_6ARG 0     // ROCm 7.2 clang-22 5-arg builtin
#endif

__device__ __forceinline__ v8f wmma16x16x32(v16h a, v16h b, v8f c) {
  // (neg_a, A, neg_b, B, c_mod, C, reuse_a, reuse_b)
  return __builtin_amdgcn_wmma_f32_16x16x32_f16(false, a, false, b, (short)0, c, false, false);
}

// A-fragment (16x32, f16) from row-major f16, row stride `ld` halves.
// Lane = row (lane&15); K: elems 0..15 -> k+off16.. (off16 = 16*(lane>>4))
__device__ __forceinline__ v16h load_a_f16(const _Float16* base, int ld, int lane, int kbase) {
  const int m = lane & 15;
  const int off8 = (lane >> 4) << 4;
  const _Float16* p = base + (size_t)m * ld + kbase + off8;
  v16h a;
#pragma unroll
  for (int j = 0; j < 8; ++j) { a[j] = p[j]; a[8 + j] = p[8 + j]; }
  return a;
}

// Same but converting from f32 source on the fly.
__device__ __forceinline__ v16h load_a_f32(const float* base, int ld, int lane, int kbase) {
  const int m = lane & 15;
  const int off8 = (lane >> 4) << 4;
  const float* p = base + (size_t)m * ld + kbase + off8;
  v16h a;
#pragma unroll
  for (int j = 0; j < 8; ++j) { a[j] = (_Float16)p[j]; a[8 + j] = (_Float16)p[8 + j]; }
  return a;
}

#if USE_TDM
// ---------------------------------------------------------------------------
// TDM: async-load a 32(k) x 16(n) f32 tile of W (row stride HIDDEN elements)
// into LDS at lds_byte_addr, laid out row-major [32][16] f32.
// Descriptor packed per CDNA5 ISA:
//   dim0 = n (contiguous, 16), dim1 = k (32 lines, stride 1024 elements).
// ---------------------------------------------------------------------------
__device__ __forceinline__ void tdm_load_w_tile(const float* gsrc, unsigned int lds_byte_addr) {
  const unsigned long long ga = (unsigned long long)(uintptr_t)gsrc;
  v4u g0;
  g0[0] = 1u;                                           // count=1 (valid user D#)
  g0[1] = lds_byte_addr;                                // lds_addr
  g0[2] = (unsigned int)(ga & 0xffffffffu);             // global_addr[31:0]
  g0[3] = (unsigned int)((ga >> 32) & 0x01ffffffu)      // global_addr[56:32]
        | (2u << 30);                                   // type=2 ("image")
  const unsigned int td0 = 16u, td1 = 32u;              // tensor dims (== tile, in-bounds)
  const unsigned int tile0 = 16u, tile1 = 32u, tile2 = 0u;
  const unsigned long long s0 = (unsigned long long)HIDDEN;  // dim0 stride (elements)
  const unsigned long long s1 = 0ull;
  v8i g1;
  g1[0] = (int)(2u << 16);                              // wg_mask=0, data_size=2 (4B)
  g1[1] = (int)((td0 & 0xffffu) << 16);                 // [63:48] tensor_dim0[15:0]
  g1[2] = (int)((td0 >> 16) | ((td1 & 0xffffu) << 16)); // td0[31:16] | td1[15:0]
  g1[3] = (int)((td1 >> 16) | (tile0 << 16));           // td1[31:16] | tile_dim0
  g1[4] = (int)(tile1 | (tile2 << 16));                 // tile_dim1 | tile_dim2
  g1[5] = (int)(s0 & 0xffffffffu);                      // dim0_stride[31:0]
  g1[6] = (int)((s0 >> 32) | ((s1 & 0xffffu) << 16));   // dim0_stride[47:32] | dim1_stride[15:0]
  g1[7] = (int)(s1 >> 16);                              // dim1_stride[47:16]
  v4i z4 = {0, 0, 0, 0};                                // groups 2/3 unused (2D tile)
#if TDM_6ARG
  v8i z8 = {0, 0, 0, 0, 0, 0, 0, 0};
  __builtin_amdgcn_tensor_load_to_lds(g0, g1, z4, z4, z8, 0);
#else
  __builtin_amdgcn_tensor_load_to_lds(g0, g1, z4, z4, 0);
#endif
}
#endif

// ---------------------------------------------------------------------------
// Stage W[k..k+31][n_base..n_base+15] (f32) into Bt[n][k] (f16, transposed)
// so each lane's B-fragment is a single contiguous 32B LDS read.
// TDM path: wave0 DMAs the f32 tile into Wf, workgroup converts/transposes.
// ---------------------------------------------------------------------------
__device__ __forceinline__ void stage_w_tile(
    const float* __restrict__ W, int k, int n_base, int wave,
    float (*Wf)[16], _Float16 (*Bt)[32]) {
#if USE_TDM
  if (wave == 0) {
    tdm_load_w_tile(W + (size_t)k * HIDDEN + n_base,
                    (unsigned int)(uintptr_t)&Wf[0][0]);
    __builtin_amdgcn_s_wait_tensorcnt(0);
  }
  __syncthreads();
  for (int i = threadIdx.x; i < 512; i += 128) {
    int kk = i >> 4, nn = i & 15;
    Bt[nn][kk] = (_Float16)Wf[kk][nn];
  }
#else
  (void)Wf;
  for (int i = threadIdx.x; i < 512; i += 128) {
    int kk = i >> 4, nn = i & 15;
    Bt[nn][kk] = (_Float16)W[(size_t)(k + kk) * HIDDEN + n_base + nn];
  }
#endif
  __syncthreads();
}

// ---------------------------------------------------------------------------
// Projection: out(f16, head-split w/ strides) = alpha * (X(f32) @ W(f32) + b)
// Block = 4 waves; wave w owns rows [bx*64 + 16w, +16), cols [by*16, +16).
// Lper = 1 << lshift (1024 or 2048).
// ---------------------------------------------------------------------------
__global__ __launch_bounds__(128) void proj_qkv_kernel(
    const float* __restrict__ X, const float* __restrict__ W,
    const float* __restrict__ bias, _Float16* __restrict__ out,
    int lshift, float alpha, int ldl, int ldd) {
  __shared__ __align__(64) float    Wf[32][16];
  __shared__ __align__(64) _Float16 Bt[16][32];
  const int lane = threadIdx.x & 31;
  const int wave = threadIdx.x >> 5;
  const int row_base = blockIdx.x * 64 + wave * 16;
  const int n_base = blockIdx.y * 16;
  const int m = lane & 15;
  const int off16 = (lane >> 4) << 4;
  const float* Xbase = X + (size_t)row_base * HIDDEN;
  v8f acc = {};
  for (int k = 0; k < HIDDEN; k += 32) {
    stage_w_tile(W, k, n_base, wave, Wf, Bt);
    v16h a = load_a_f32(Xbase, HIDDEN, lane, k);
    v16h b = *(const v16h*)(&Bt[m][0] + off16);
    acc = wmma16x16x32(a, b, acc);
    __syncthreads();
  }
  const int hi8 = (lane >> 4) << 3;
#pragma unroll
  for (int r = 0; r < 8; ++r) {
    int gm = row_base + hi8 + r;         // C/D layout: lanes<16 -> M=r, else M=8+r
    int gn = n_base + m;                 // N = lane&15
    float v = (acc[r] + bias[gn]) * alpha;
    int bidx = gm >> lshift, l = gm & ((1 << lshift) - 1);
    int hh = gn >> 6, dd = gn & 63;
    size_t addr = ((size_t)(bidx * HEADS + hh) << lshift) * DK
                + (size_t)l * (size_t)ldl + (size_t)dd * (size_t)ldd;
    out[addr] = (_Float16)v;
  }
}

// ---------------------------------------------------------------------------
// Flash attention: one wave = 16 q rows of one (b,h); kv chunks of 64.
// Q [bh][LQ][64] f16 (pre-scaled), K [bh][LKV][64] f16, V transposed
// [bh][64][LKV] f16 so every B-fragment is a contiguous 32B load.
// ---------------------------------------------------------------------------
__global__ __launch_bounds__(128) void attention_kernel(
    const _Float16* __restrict__ Qh, const _Float16* __restrict__ Kh,
    const _Float16* __restrict__ Vt, const unsigned char* __restrict__ mask,
    _Float16* __restrict__ ctx) {
  __shared__ __align__(64) _Float16 Psm[4][16 * 64];
  const int lane = threadIdx.x & 31;
  const int wave = threadIdx.x >> 5;
  const int bh = blockIdx.x;                 // b*HEADS + h
  const int b = bh >> 4, h = bh & 15;
  const int q_base = blockIdx.y * 64 + wave * 16;
  const int n = lane & 15;
  const int off16 = (lane >> 4) << 4;
  const int hi8 = (lane >> 4) << 3;
  _Float16* Pw = Psm[wave];

  const _Float16* Qp = Qh + ((size_t)bh * LQ_ + q_base) * DK;
  const _Float16* Kp = Kh + (size_t)bh * LKV_ * DK;
  const _Float16* Vp = Vt + (size_t)bh * DK * LKV_;
  const unsigned char* Mp = mask + ((size_t)b * LQ_ + q_base) * LKV_;

  const v16h aq0 = load_a_f16(Qp, DK, lane, 0);    // loop-invariant Q fragments
  const v16h aq1 = load_a_f16(Qp, DK, lane, 32);

  v8f o0 = {}, o1 = {}, o2 = {}, o3 = {};
  float mrun[8], lrun[8];
#pragma unroll
  for (int r = 0; r < 8; ++r) { mrun[r] = -3.0e38f; lrun[r] = 0.f; }

  for (int kv = 0; kv < LKV_; kv += 64) {
    if (kv + 64 < LKV_) {                          // global_prefetch_b8 next chunk
      __builtin_prefetch(Kp + (size_t)(kv + 64 + lane) * DK, 0, 1);
      __builtin_prefetch(Vp + (size_t)lane * LKV_ + kv + 64, 0, 1);
    }
    // S = Q K^T for 4 column tiles (contraction d=64 -> 2 WMMAs each)
    v8f s[4];
#pragma unroll
    for (int c = 0; c < 4; ++c) {
      const _Float16* krow = Kp + (size_t)(kv + c * 16 + n) * DK;
      v8f a = {};
      a = wmma16x16x32(aq0, *(const v16h*)(krow + off16), a);
      a = wmma16x16x32(aq1, *(const v16h*)(krow + 32 + off16), a);
      s[c] = a;
    }
    // mask + online softmax (row M=hi8+r is spread across this 16-lane half)
#pragma unroll
    for (int r = 0; r < 8; ++r) {
      const unsigned char* mrow = Mp + (size_t)(hi8 + r) * LKV_ + kv + n;
#pragma unroll
      for (int c = 0; c < 4; ++c)
        if (mrow[c * 16]) s[c][r] = -1.0e18f;      // reference NEG_INF
      float mx = fmaxf(fmaxf(s[0][r], s[1][r]), fmaxf(s[2][r], s[3][r]));
#pragma unroll
      for (int d = 8; d >= 1; d >>= 1) mx = fmaxf(mx, __shfl_xor(mx, d, 32));
      float mnew = fmaxf(mrun[r], mx);
      float scale = __expf(mrun[r] - mnew);
      float rowsum = 0.f;
#pragma unroll
      for (int c = 0; c < 4; ++c) {
        float p = __expf(s[c][r] - mnew);
        s[c][r] = p;
        rowsum += p;
      }
#pragma unroll
      for (int d = 8; d >= 1; d >>= 1) rowsum += __shfl_xor(rowsum, d, 32);
      lrun[r] = lrun[r] * scale + rowsum;
      mrun[r] = mnew;
      o0[r] *= scale; o1[r] *= scale; o2[r] *= scale; o3[r] *= scale;
    }
    // Stage P (C-layout) -> LDS row-major f16 so it can re-enter as A-layout.
    // Wave-private region; same-wave DS ops are in-order (DScnt).
#pragma unroll
    for (int c = 0; c < 4; ++c)
#pragma unroll
      for (int r = 0; r < 8; ++r)
        Pw[(hi8 + r) * 64 + c * 16 + n] = (_Float16)s[c][r];
    v16h ap0 = load_a_f16(Pw, 64, lane, 0);
    v16h ap1 = load_a_f16(Pw, 64, lane, 32);
    // O += P @ V  (4 d-tiles x 2 k-steps)
    const _Float16* v0 = Vp + (size_t)(n) * LKV_ + kv;
    const _Float16* v1 = Vp + (size_t)(16 + n) * LKV_ + kv;
    const _Float16* v2 = Vp + (size_t)(32 + n) * LKV_ + kv;
    const _Float16* v3 = Vp + (size_t)(48 + n) * LKV_ + kv;
    o0 = wmma16x16x32(ap0, *(const v16h*)(v0 + off16), o0);
    o0 = wmma16x16x32(ap1, *(const v16h*)(v0 + 32 + off16), o0);
    o1 = wmma16x16x32(ap0, *(const v16h*)(v1 + off16), o1);
    o1 = wmma16x16x32(ap1, *(const v16h*)(v1 + 32 + off16), o1);
    o2 = wmma16x16x32(ap0, *(const v16h*)(v2 + off16), o2);
    o2 = wmma16x16x32(ap1, *(const v16h*)(v2 + 32 + off16), o2);
    o3 = wmma16x16x32(ap0, *(const v16h*)(v3 + off16), o3);
    o3 = wmma16x16x32(ap1, *(const v16h*)(v3 + 32 + off16), o3);
  }
  // normalize and store ctx (merged heads, f16) [B, LQ, HIDDEN]
#pragma unroll
  for (int r = 0; r < 8; ++r) {
    float rl = 1.0f / lrun[r];
    size_t base = ((size_t)b * LQ_ + q_base + hi8 + r) * HIDDEN + (size_t)h * DK + n;
    ctx[base +  0] = (_Float16)(o0[r] * rl);
    ctx[base + 16] = (_Float16)(o1[r] * rl);
    ctx[base + 32] = (_Float16)(o2[r] * rl);
    ctx[base + 48] = (_Float16)(o3[r] * rl);
  }
}

// ---------------------------------------------------------------------------
// Output projection: out(f32) = ctx(f16) @ Wo + bo
// ---------------------------------------------------------------------------
__global__ __launch_bounds__(128) void out_proj_kernel(
    const _Float16* __restrict__ Xh, const float* __restrict__ W,
    const float* __restrict__ bias, float* __restrict__ out) {
  __shared__ __align__(64) float    Wf[32][16];
  __shared__ __align__(64) _Float16 Bt[16][32];
  const int lane = threadIdx.x & 31;
  const int wave = threadIdx.x >> 5;
  const int row_base = blockIdx.x * 64 + wave * 16;
  const int n_base = blockIdx.y * 16;
  const int m = lane & 15;
  const int off16 = (lane >> 4) << 4;
  const _Float16* Xbase = Xh + (size_t)row_base * HIDDEN;
  v8f acc = {};
  for (int k = 0; k < HIDDEN; k += 32) {
    stage_w_tile(W, k, n_base, wave, Wf, Bt);
    v16h a = load_a_f16(Xbase, HIDDEN, lane, k);
    v16h b = *(const v16h*)(&Bt[m][0] + off16);
    acc = wmma16x16x32(a, b, acc);
    __syncthreads();
  }
  const int hi8 = (lane >> 4) << 3;
#pragma unroll
  for (int r = 0; r < 8; ++r) {
    int gm = row_base + hi8 + r;
    int gn = n_base + m;
    out[(size_t)gm * HIDDEN + gn] = acc[r] + bias[gn];
  }
}

// ---------------------------------------------------------------------------
extern "C" void kernel_launch(void* const* d_in, const int* in_sizes, int n_in,
                              void* d_out, int out_size, void* d_ws, size_t ws_size,
                              hipStream_t stream) {
  (void)in_sizes; (void)n_in; (void)out_size; (void)ws_size;
  const float* tgt    = (const float*)d_in[0];
  const float* memory = (const float*)d_in[1];
  const unsigned char* mask = (const unsigned char*)d_in[2];
  const float* Wq = (const float*)d_in[3];
  const float* bq = (const float*)d_in[4];
  const float* Wk = (const float*)d_in[5];
  const float* bk = (const float*)d_in[6];
  const float* Wv = (const float*)d_in[7];
  const float* bv = (const float*)d_in[8];
  const float* Wo = (const float*)d_in[9];
  const float* bo = (const float*)d_in[10];
  float* out = (float*)d_out;

  // workspace layout (f16): Qh 8MB | Kh 16MB | Vt 16MB | ctx 8MB = 48MB
  char* ws = (char*)d_ws;
  _Float16* Qh  = (_Float16*)(ws);
  _Float16* Kh  = (_Float16*)(ws + (size_t)8  * 1024 * 1024);
  _Float16* Vt  = (_Float16*)(ws + (size_t)24 * 1024 * 1024);
  _Float16* ctx = (_Float16*)(ws + (size_t)40 * 1024 * 1024);

  dim3 blk(128);
  // Q = (tgt@Wq + bq) * 1/sqrt(64), layout [b,h,l,d]  (Lper = 1<<10)
  hipLaunchKernelGGL(proj_qkv_kernel, dim3(64, 64), blk, 0, stream,
                     tgt, Wq, bq, Qh, 10, 0.125f, DK, 1);
  // K = memory@Wk + bk, layout [b,h,kv,d]              (Lper = 1<<11)
  hipLaunchKernelGGL(proj_qkv_kernel, dim3(128, 64), blk, 0, stream,
                     memory, Wk, bk, Kh, 11, 1.0f, DK, 1);
  // V = memory@Wv + bv, TRANSPOSED layout [b,h,d,kv]
  hipLaunchKernelGGL(proj_qkv_kernel, dim3(128, 64), blk, 0, stream,
                     memory, Wv, bv, Vt, 11, 1.0f, 1, LKV_);
  // flash attention -> ctx [b, lq, h*64+d] f16
  hipLaunchKernelGGL(attention_kernel, dim3(64, 16), blk, 0, stream,
                     Qh, Kh, Vt, mask, ctx);
  // out = ctx@Wo + bo (f32)
  hipLaunchKernelGGL(out_proj_kernel, dim3(64, 64), blk, 0, stream,
                     ctx, Wo, bo, out);
}
